// KoLeoLoss_70712341562212
// MI455X (gfx1250) — compile-verified, matrix operand order checked by reading, not run on hardware
//
#include <hip/hip_runtime.h>
#include <hip/hip_bf16.h>

typedef _Float16 v4h  __attribute__((ext_vector_type(4)));
typedef _Float16 v8h  __attribute__((ext_vector_type(8)));
typedef _Float16 v16h __attribute__((ext_vector_type(16)));
typedef float    v8f  __attribute__((ext_vector_type(8)));

#define DIMS 1024
#define ROWS_PER_BLOCK 128
#define WAVES_PER_BLOCK 8
#define NORM_EPS  1e-8f
#define PDIST_EPS 1e-6f

// ---------------------------------------------------------------------------
// Kernel 1: per-row L2 normalize, f32 -> f16   (one block per row)
// ---------------------------------------------------------------------------
__global__ __launch_bounds__(256) void koleo_normalize(const float* __restrict__ x,
                                                       _Float16* __restrict__ xh) {
  const int row = blockIdx.x;
  const int tid = threadIdx.x;
  const float4 v = ((const float4*)(x + (size_t)row * DIMS))[tid];
  float s = v.x * v.x + v.y * v.y + v.z * v.z + v.w * v.w;
  #pragma unroll
  for (int off = 16; off > 0; off >>= 1) s += __shfl_xor(s, off, 32);
  __shared__ float red[8];
  const int wave = tid >> 5, lane = tid & 31;
  if (lane == 0) red[wave] = s;
  __syncthreads();
  if (tid == 0) {
    float t = 0.f;
    #pragma unroll
    for (int i = 0; i < 8; ++i) t += red[i];
    red[0] = t;
  }
  __syncthreads();
  const float inv = 1.0f / fmaxf(sqrtf(red[0]), NORM_EPS);
  v4h h;
  h[0] = (_Float16)(v.x * inv);
  h[1] = (_Float16)(v.y * inv);
  h[2] = (_Float16)(v.z * inv);
  h[3] = (_Float16)(v.w * inv);
  ((v4h*)(xh + (size_t)row * DIMS))[tid] = h;
}

// ---------------------------------------------------------------------------
// Kernel 2: NN argmax over cosine similarity via f16 WMMA.
// Block stages 128 A-rows in LDS (256 KB). The 8 waves form a 4 (row-group)
// x 2 (column-group) grid. Each wave computes a 32x32 output patch per pass
// (2x2 register blocking = 4 independent v_wmma_f32_16x16x32_f16 chains):
// each A fragment is reused across 2 column tiles and each B fragment across
// 2 row sub-tiles -> 0.5 KB LDS + 0.5 KB global per WMMA. Column tiles are
// split across the 2 column-groups (halves L2 traffic vs. all-waves-all-
// columns); their per-row (max, argmax) results merge through LDS at the end.
// ---------------------------------------------------------------------------
__global__ __launch_bounds__(WAVES_PER_BLOCK * 32) void koleo_argmax_nn(
    const _Float16* __restrict__ xh, int* __restrict__ nn, int N) {
  __shared__ _Float16 As[ROWS_PER_BLOCK * DIMS];  // 256 KB
  __shared__ float sval[ROWS_PER_BLOCK];
  __shared__ int   sidx[ROWS_PER_BLOCK];
  const int tid      = threadIdx.x;
  const int rowBase0 = blockIdx.x * ROWS_PER_BLOCK;

  // Cooperative stage of this block's 128 A-rows into LDS (16B vectors).
  {
    const v8h* src = (const v8h*)(xh + (size_t)rowBase0 * DIMS);
    v8h*       dst = (v8h*)As;
    const int nvec = ROWS_PER_BLOCK * DIMS / 8;
    for (int i = tid; i < nvec; i += WAVES_PER_BLOCK * 32) dst[i] = src[i];
  }
  __syncthreads();

  const int wave  = tid >> 5;
  const int lane  = tid & 31;
  const int ln    = lane & 15;   // A: M row / B: N col / C: N col
  const int khalf = lane >> 4;   // which K-half this lane holds
  const int rg    = wave >> 1;   // row group: rows [rg*32, rg*32+32)
  const int cg    = wave & 1;    // column group: passes i = cg, cg+2, ...

  const _Float16* aRow0 = As + (size_t)(rg * 32 + ln) * DIMS + khalf * 8;
  const _Float16* aRow1 = aRow0 + (size_t)16 * DIMS;

  float bv[2][8];
  int   bi[2][8];
  #pragma unroll
  for (int s = 0; s < 2; ++s)
    #pragma unroll
    for (int v = 0; v < 8; ++v) { bv[s][v] = -3.0e38f; bi[s][v] = 0; }

  const int npass = N >> 5;  // 32 columns per pass
  for (int i = cg; i < npass; i += 2) {
    const int colBase = i << 5;
    // Symmetry: B fragment (32x16 K-major) of columns [colBase,+16) has the
    // same per-lane k layout (8*khalf+s, then 16+8*khalf+s-8) as an A fragment of those rows.
    const _Float16* bRow0 = xh + (size_t)(colBase + ln) * DIMS + khalf * 8;
    const _Float16* bRow1 = bRow0 + (size_t)16 * DIMS;

    v8f c00 = {0.f, 0.f, 0.f, 0.f, 0.f, 0.f, 0.f, 0.f};
    v8f c01 = c00, c10 = c00, c11 = c00;
    #pragma unroll 4
    for (int kc = 0; kc < 32; ++kc) {
      const int kb = kc * 32;
      v16h a0 = __builtin_shufflevector(*(const v8h*)(aRow0 + kb),
                                        *(const v8h*)(aRow0 + kb + 16),
                                        0, 1, 2, 3, 4, 5, 6, 7,
                                        8, 9, 10, 11, 12, 13, 14, 15);
      v16h a1 = __builtin_shufflevector(*(const v8h*)(aRow1 + kb),
                                        *(const v8h*)(aRow1 + kb + 16),
                                        0, 1, 2, 3, 4, 5, 6, 7,
                                        8, 9, 10, 11, 12, 13, 14, 15);
      v16h b0 = __builtin_shufflevector(*(const v8h*)(bRow0 + kb),
                                        *(const v8h*)(bRow0 + kb + 16),
                                        0, 1, 2, 3, 4, 5, 6, 7,
                                        8, 9, 10, 11, 12, 13, 14, 15);
      v16h b1 = __builtin_shufflevector(*(const v8h*)(bRow1 + kb),
                                        *(const v8h*)(bRow1 + kb + 16),
                                        0, 1, 2, 3, 4, 5, 6, 7,
                                        8, 9, 10, 11, 12, 13, 14, 15);
      c00 = __builtin_amdgcn_wmma_f32_16x16x32_f16(false, a0, false, b0,
                                                   (short)0, c00, false, false);
      c01 = __builtin_amdgcn_wmma_f32_16x16x32_f16(false, a0, false, b1,
                                                   (short)0, c01, false, false);
      c10 = __builtin_amdgcn_wmma_f32_16x16x32_f16(false, a1, false, b0,
                                                   (short)0, c10, false, false);
      c11 = __builtin_amdgcn_wmma_f32_16x16x32_f16(false, a1, false, b1,
                                                   (short)0, c11, false, false);
    }

    // C layout: lane ln -> N=ln ; VGPR v -> M = v + 8*khalf.
    const int ncol0 = colBase + ln;
    const int ncol1 = colBase + 16 + ln;
    #pragma unroll
    for (int v = 0; v < 8; ++v) {
      const int mrow0 = rowBase0 + rg * 32 + v + 8 * khalf;
      const int mrow1 = mrow0 + 16;
      const float v00 = c00[v], v01 = c01[v];
      const float v10 = c10[v], v11 = c11[v];
      if (mrow0 != ncol0 && v00 > bv[0][v]) { bv[0][v] = v00; bi[0][v] = ncol0; }
      if (mrow0 != ncol1 && v01 > bv[0][v]) { bv[0][v] = v01; bi[0][v] = ncol1; }
      if (mrow1 != ncol0 && v10 > bv[1][v]) { bv[1][v] = v10; bi[1][v] = ncol0; }
      if (mrow1 != ncol1 && v11 > bv[1][v]) { bv[1][v] = v11; bi[1][v] = ncol1; }
    }
  }

  // Reduce (max, argmax) across the 16 lanes of each khalf group, then merge
  // the two column-groups' candidates through LDS.
  float rval[2];
  int   ridx[2];
  #pragma unroll
  for (int s = 0; s < 2; ++s) {
    #pragma unroll
    for (int v = 0; v < 8; ++v) {
      float val = bv[s][v];
      int   idx = bi[s][v];
      #pragma unroll
      for (int off = 8; off > 0; off >>= 1) {
        const float ov = __shfl_xor(val, off, 32);
        const int   oi = __shfl_xor(idx, off, 32);
        if (ov > val || (ov == val && oi < idx)) { val = ov; idx = oi; }
      }
      if (v == 0) { rval[s] = val; ridx[s] = idx; }
      else {
        // store per-v winner immediately below (only lane ln==0 is valid);
        // handled via LDS writes inside the loop for simplicity:
      }
      const int rowLoc = rg * 32 + s * 16 + v + 8 * khalf;
      if (ln == 0 && cg == 0) { sval[rowLoc] = val; sidx[rowLoc] = idx; }
      bv[s][v] = val;  // keep reduced winner for phase 2
      bi[s][v] = idx;
    }
  }
  __syncthreads();
  #pragma unroll
  for (int s = 0; s < 2; ++s) {
    #pragma unroll
    for (int v = 0; v < 8; ++v) {
      const int rowLoc = rg * 32 + s * 16 + v + 8 * khalf;
      if (ln == 0 && cg == 1) {
        float val = bv[s][v];
        int   idx = bi[s][v];
        const float ov = sval[rowLoc];
        const int   oi = sidx[rowLoc];
        if (ov > val || (ov == val && oi < idx)) { val = ov; idx = oi; }
        nn[rowBase0 + rowLoc] = idx;
      }
    }
  }
}

// ---------------------------------------------------------------------------
// Kernel 3: per-row log distance to NN in f32 (one block per row)
// ---------------------------------------------------------------------------
__global__ __launch_bounds__(256) void koleo_dist(const float* __restrict__ x,
                                                  const int* __restrict__ nn,
                                                  float* __restrict__ logs) {
  const int row = blockIdx.x;
  const int j   = nn[row];
  const int tid = threadIdx.x;
  const float4 va = ((const float4*)(x + (size_t)row * DIMS))[tid];
  const float4 vb = ((const float4*)(x + (size_t)j * DIMS))[tid];
  const float d0 = va.x - vb.x + PDIST_EPS;
  const float d1 = va.y - vb.y + PDIST_EPS;
  const float d2 = va.z - vb.z + PDIST_EPS;
  const float d3 = va.w - vb.w + PDIST_EPS;
  float s = d0 * d0 + d1 * d1 + d2 * d2 + d3 * d3;
  #pragma unroll
  for (int off = 16; off > 0; off >>= 1) s += __shfl_xor(s, off, 32);
  __shared__ float red[8];
  const int wave = tid >> 5, lane = tid & 31;
  if (lane == 0) red[wave] = s;
  __syncthreads();
  if (tid == 0) {
    float t = 0.f;
    #pragma unroll
    for (int i = 0; i < 8; ++i) t += red[i];
    logs[row] = 0.5f * logf(t);  // log(sqrt(sumsq))
  }
}

// ---------------------------------------------------------------------------
// Kernel 4: deterministic fixed-order final reduction  loss = -mean(logs)
// ---------------------------------------------------------------------------
__global__ __launch_bounds__(256) void koleo_reduce(const float* __restrict__ logs,
                                                    float* __restrict__ out, int N) {
  const int tid = threadIdx.x;
  float s = 0.f;
  for (int i = tid; i < N; i += 256) s += logs[i];
  #pragma unroll
  for (int off = 16; off > 0; off >>= 1) s += __shfl_xor(s, off, 32);
  __shared__ float red[8];
  const int wave = tid >> 5, lane = tid & 31;
  if (lane == 0) red[wave] = s;
  __syncthreads();
  if (tid == 0) {
    float t = 0.f;
    #pragma unroll
    for (int i = 0; i < 8; ++i) t += red[i];
    out[0] = -t / (float)N;
  }
}

// ---------------------------------------------------------------------------
extern "C" void kernel_launch(void* const* d_in, const int* in_sizes, int n_in,
                              void* d_out, int out_size, void* d_ws, size_t ws_size,
                              hipStream_t stream) {
  const float* x = (const float*)d_in[0];
  const int N = in_sizes[0] / DIMS;  // 16384

  // Workspace layout: [ xh : N*D f16 ][ nn : N int ][ logs : N float ]
  _Float16* xh   = (_Float16*)d_ws;
  int*      nn   = (int*)((char*)d_ws + (size_t)N * DIMS * sizeof(_Float16));
  float*    logs = (float*)((char*)nn + (size_t)N * sizeof(int));
  float*    out  = (float*)d_out;

  koleo_normalize<<<N, 256, 0, stream>>>(x, xh);
  koleo_argmax_nn<<<N / ROWS_PER_BLOCK, WAVES_PER_BLOCK * 32, 0, stream>>>(xh, nn, N);
  koleo_dist<<<N, 256, 0, stream>>>(x, nn, logs);
  koleo_reduce<<<1, 256, 0, stream>>>(logs, out, N);
}
